// SSMEvaluator2_33406255629020
// MI455X (gfx1250) — matrix operand order repaired and checked by reading, not compiled, not run on hardware
//
#include <hip/hip_runtime.h>
#include <hip/hip_bf16.h>
#include <math.h>

typedef __attribute__((ext_vector_type(16))) _Float16 v16h;
typedef __attribute__((ext_vector_type(8)))  _Float16 v8h;
typedef __attribute__((ext_vector_type(8)))  float    v8f;

#define SEQ    256
#define BATCH  64
#define SREP   128
#define ROWS   8192          // SREP * BATCH
#define INED   32
#define ZD     64
#define HID    128
#define NC     8             // NCUR
#define RPB    64            // rows per block (4 wave32 slabs of 16 rows)
#define NTHR   128
#define NBLK   (ROWS / RPB)  // 128
#define LOG2PI_F 1.8378770664093453f

// ---- LDS layout (dynamic). Offsets in _Float16 units for the half region. ----
#define OWH   0                          // W_h^T  [128][96]
#define OWG   (OWH  + 128*96)            // W_g^T  [64][128]
#define OWM   (OWG  + 64*128)            // W_m^T  [64][128]
#define OWS   (OWM  + 64*128)            // W_s^T  [64][128]
#define OWZ   (OWS  + 64*128)            // W_z^T  [64][64]
#define OWE1  (OWZ  + 64*64)             // W_e1^T [128][64]
#define OWE2  (OWE1 + 128*64)            // W_e2^T [32][128]
#define OA    (OWE2 + 32*128)            // A = [x | z]  [RPB][96]
#define OHB   (OA   + RPB*96)            // h / he buffer [RPB][128]
#define NHALF (OHB  + RPB*128)
// float region (offsets in floats)
#define FBH   0
#define FBE1  128
#define FBG   256
#define FBM   320
#define FBZ   384
#define FBS   448
#define FSH   512     // iaf_shift
#define FLS   576     // iaf_logscale (raw)
#define FEL   640     // exp(iaf_logscale)
#define FBE2  704
#define NFLT  736
#define SMEM_BYTES ((size_t)NHALF * 2 + (size_t)NFLT * 4)   // 138112 B

// ---------------- math helpers ----------------
__device__ __forceinline__ float sigmoidf_(float x) { return 1.0f / (1.0f + __expf(-x)); }
__device__ __forceinline__ float softplusf_(float x) {
  return fmaxf(x, 0.0f) + __logf(1.0f + __expf(-fabsf(x)));
}
__device__ __forceinline__ unsigned hashu_(unsigned x) {
  x ^= x >> 16; x *= 0x7feb352dU; x ^= x >> 15; x *= 0x846ca68bU; x ^= x >> 16;
  return x;
}
__device__ __forceinline__ float unif01_(unsigned s) {
  return ((float)(hashu_(s) >> 8) + 0.5f) * (1.0f / 16777216.0f);
}
__device__ __forceinline__ float normal_(unsigned s) {
  float u1 = unif01_(s ^ 0x68bc21ebU);
  float u2 = unif01_(s * 0x9E3779B9U + 0x7F4A7C15U);
  return __fsqrt_rn(fmaxf(-2.0f * __logf(u1), 0.0f)) * __cosf(6.283185307f * u2);
}

// ---- WMMA fragment loaders (wave32, V_WMMA_F32_16X16X32_F16), transposed-GEMM roles ----
// A-operand (16x32 f16) from a TRANSPOSED weight tile W^T[n][k], base = W^T + nOff*ldk.
// Lane: m = lane&15 (output-feature row of the tile); per lane K = {kb..kb+7, kb+16..kb+23}, kb = 8*(lane>=16).
__device__ __forceinline__ v16h load_A(const _Float16* __restrict__ base, int ldk, int kOff) {
  const int lane = threadIdx.x & 31;
  const int m    = lane & 15;
  const int kb   = (lane >> 4) << 3;             // 0 or 8
  const _Float16* p = base + m * ldk + kOff + kb;
  v8h lo = *(const v8h*)(p);                     // K = kb .. kb+7
  v8h hi = *(const v8h*)(p + 16);                // K = kb+16 .. kb+23
  v16h a;
#pragma unroll
  for (int i = 0; i < 8; ++i) { a[i] = lo[i]; a[8 + i] = hi[i]; }
  return a;
}
// B-operand (32x16 f16) = activation^T, read from ROW-MAJOR activations act[row][k].
// Lane: n = lane&15 (activation row); per lane K = {kb..kb+7, kb+16..kb+23}, kb = 8*(lane>=16).
__device__ __forceinline__ v16h load_B(const _Float16* __restrict__ act, int ldk, int kOff) {
  const int lane = threadIdx.x & 31;
  const int n  = lane & 15;
  const int kb = (lane >> 4) << 3;               // 0 or 8
  const _Float16* p = act + n * ldk + kOff + kb;
  v8h lo = *(const v8h*)(p);
  v8h hi = *(const v8h*)(p + 16);
  v16h b;
#pragma unroll
  for (int i = 0; i < 8; ++i) { b[i] = lo[i]; b[8 + i] = hi[i]; }
  return b;
}
__device__ __forceinline__ v8f wmma_f16(v16h a, v16h b, v8f c) {
  return __builtin_amdgcn_wmma_f32_16x16x32_f16(false, a, false, b, (short)0, c, false, false);
}

// transpose-load a KxN row-major f32 weight into N x K f16 LDS
__device__ __forceinline__ void load_wT(const float* __restrict__ W, _Float16* __restrict__ WT,
                                        int K, int N, int tid) {
  for (int i = tid; i < K * N; i += NTHR) {
    int k = i / N, n = i - k * N;                // N is a power of two here
    WT[n * K + k] = (_Float16)W[i];
  }
}

__global__ __launch_bounds__(NTHR)
void ssm_persistent_kernel(const float* __restrict__ xin,
                           const float* __restrict__ Wh,  const float* __restrict__ bh,
                           const float* __restrict__ Wg,  const float* __restrict__ bg,
                           const float* __restrict__ Wm,  const float* __restrict__ bm,
                           const float* __restrict__ Wz,  const float* __restrict__ bz,
                           const float* __restrict__ Ws,  const float* __restrict__ bs,
                           const float* __restrict__ We1, const float* __restrict__ be1,
                           const float* __restrict__ We2, const float* __restrict__ be2,
                           const float* __restrict__ iafShift,
                           const float* __restrict__ iafLogscale,
                           float* __restrict__ out) {
  extern __shared__ __align__(16) unsigned char smemRaw[];
  _Float16* sHalf = (_Float16*)smemRaw;
  float*    sF    = (float*)(sHalf + NHALF);

  _Float16* sWh  = sHalf + OWH;
  _Float16* sWg  = sHalf + OWG;
  _Float16* sWm  = sHalf + OWM;
  _Float16* sWs  = sHalf + OWS;
  _Float16* sWz  = sHalf + OWZ;
  _Float16* sWe1 = sHalf + OWE1;
  _Float16* sWe2 = sHalf + OWE2;
  _Float16* sA   = sHalf + OA;
  _Float16* sH   = sHalf + OHB;

  const int tid = threadIdx.x;

  // ---- one-time: weights (transposed, f16), biases, IAF params, zero z-region of A ----
  load_wT(Wh,  sWh,  96, 128, tid);
  load_wT(Wg,  sWg, 128,  64, tid);
  load_wT(Wm,  sWm, 128,  64, tid);
  load_wT(Ws,  sWs, 128,  64, tid);
  load_wT(Wz,  sWz,  64,  64, tid);
  load_wT(We1, sWe1, 64, 128, tid);
  load_wT(We2, sWe2,128,  32, tid);
  if (tid < 128) { sF[FBH + tid] = bh[tid]; sF[FBE1 + tid] = be1[tid]; }
  if (tid < 64) {
    sF[FBG + tid] = bg[tid]; sF[FBM + tid] = bm[tid];
    sF[FBZ + tid] = bz[tid]; sF[FBS + tid] = bs[tid];
    sF[FSH + tid] = iafShift[tid];
    float ls = iafLogscale[tid];
    sF[FLS + tid] = ls; sF[FEL + tid] = __expf(ls);
  }
  if (tid < 32) sF[FBE2 + tid] = be2[tid];
  for (int i = tid; i < RPB * 96; i += NTHR) sA[i] = (_Float16)0.0f;  // z0 = 0
  __syncthreads();

  float lsSum = 0.0f;
#pragma unroll
  for (int i = 0; i < ZD; ++i) lsSum += sF[FLS + i];

  // ---- per-wave slab: each wave32 owns 16 rows for all 256 steps; zero barriers in loop ----
  const int wave = tid >> 5;
  const int lane = tid & 31;
  const int n16  = lane & 15;          // activation row within slab (accumulator N dim)
  const int hh   = lane >> 4;          // feature-half select (accumulator M half)
  const int rbase = wave * 16;
  const _Float16* aSlab = sA + rbase * 96;
  _Float16*       hSlab = sH + rbase * HID;
  const int growWave = blockIdx.x * RPB + rbase;   // global row of slab row 0
  const int grow     = growWave + n16;             // this lane's global row

  // output sections (flat, in reference return order)
  float* outS = out;                               // samples   (SEQ,S,B,NC,3)
  float* outF = out + (size_t)50331648;            // fractions (SEQ,S,B,NC)
  float* outX = out + (size_t)67108864;            // x_lps     (SEQ,S,B,NC,3)
  float* outZ = out + (size_t)117440512;           // z_lps     (SEQ,S,B)

  for (int t = 0; t < SEQ; ++t) {
    // (a) fill x into A cols [0,32): lane -> row = lane&15, 16 cols starting at 16*hh
    {
      const int row = n16;
      const int b   = (growWave + row) & (BATCH - 1);
      const float* xp = xin + ((size_t)t * BATCH + b) * INED + hh * 16;
      _Float16*    ap = sA + (rbase + row) * 96 + hh * 16;
#pragma unroll
      for (int i = 0; i < 16; ++i) ap[i] = (_Float16)xp[i];
    }

    // (b) h^T = W_h^T @ [x|z]^T  (i.e. h = relu([x|z]W_h + b_h)), M=feat N=rows K=96
    v16h b0 = load_B(aSlab, 96, 0);
    v16h b1 = load_B(aSlab, 96, 32);   // z_prev cols 32..63 (reused as B for W_z GEMM)
    v16h b2 = load_B(aSlab, 96, 64);   // z_prev cols 64..95
#pragma unroll
    for (int nt = 0; nt < 8; ++nt) {
      v8f acc = {};
      acc = wmma_f16(load_A(sWh + nt * 16 * 96, 96, 0),  b0, acc);
      acc = wmma_f16(load_A(sWh + nt * 16 * 96, 96, 32), b1, acc);
      acc = wmma_f16(load_A(sWh + nt * 16 * 96, 96, 64), b2, acc);
      const float* bp = sF + FBH + nt * 16 + 8 * hh;   // 8 consecutive bias floats
      v8h hv;
#pragma unroll
      for (int r = 0; r < 8; ++r) hv[r] = (_Float16)fmaxf(acc[r] + bp[r], 0.0f);
      *(v8h*)(hSlab + n16 * HID + nt * 16 + 8 * hh) = hv;   // one ds_store_b128
    }

    // (c,d) heads g/mu/s off h (K=128) + zlin off z_prev (K=64); fused IAF + RNG + z_lp
    v16h hb[4];
#pragma unroll
    for (int kt = 0; kt < 4; ++kt) hb[kt] = load_B(hSlab, HID, kt * 32);

    float lpSum = 0.0f;
#pragma unroll
    for (int nt = 0; nt < 4; ++nt) {
      v8f ag = {}, am = {}, as = {}, az = {};
#pragma unroll
      for (int kt = 0; kt < 4; ++kt) {
        ag = wmma_f16(load_A(sWg + nt * 16 * 128, 128, kt * 32), hb[kt], ag);
        am = wmma_f16(load_A(sWm + nt * 16 * 128, 128, kt * 32), hb[kt], am);
        as = wmma_f16(load_A(sWs + nt * 16 * 128, 128, kt * 32), hb[kt], as);
      }
      az = wmma_f16(load_A(sWz + nt * 16 * 64, 64, 0),  b1, az);
      az = wmma_f16(load_A(sWz + nt * 16 * 64, 64, 32), b2, az);

      const int f0 = nt * 16 + 8 * hh;                 // first of 8 consecutive features
      const float* pG  = sF + FBG + f0;
      const float* pM  = sF + FBM + f0;
      const float* pZ  = sF + FBZ + f0;
      const float* pS  = sF + FBS + f0;
      const float* pSh = sF + FSH + f0;
      const float* pEl = sF + FEL + f0;
      v8h zv;
#pragma unroll
      for (int r = 0; r < 8; ++r) {
        const float g   = sigmoidf_(ag[r] + pG[r]);
        const float mu  = am[r] + pM[r];
        const float zl  = az[r] + pZ[r];
        const float sc  = softplusf_(as[r] + pS[r]) + 1.0e-4f;
        const float loc = (1.0f - g) * zl + g * mu;
        const unsigned seed = ((unsigned)(t * ROWS + grow) << 7) | ((unsigned)(f0 + r) << 1);
        const float eps = normal_(seed);
        const float z   = pSh[r] + pEl[r] * (loc + sc * eps);
        lpSum += -0.5f * eps * eps - __logf(sc);
        zv[r] = (_Float16)z;
      }
      *(v8h*)(sA + (rbase + n16) * 96 + 32 + f0) = zv;   // z carry: one ds_store_b128
    }

    // (e) z_lp: in-lane sum covers this feature-half; one xor-16 combine, coalesced store
    {
      const float v = lpSum + __shfl_xor(lpSum, 16, 32);
      if (hh == 0)
        outZ[(size_t)t * ROWS + grow] = v - 32.0f * LOG2PI_F - lsSum;
    }

    // (f) he^T = W_e1^T @ z^T (he = relu(z W_e1 + b_e1)), K=64
    v16h zb0 = load_B(aSlab, 96, 32);
    v16h zb1 = load_B(aSlab, 96, 64);
#pragma unroll
    for (int nt = 0; nt < 8; ++nt) {
      v8f acc = {};
      acc = wmma_f16(load_A(sWe1 + nt * 16 * 64, 64, 0),  zb0, acc);
      acc = wmma_f16(load_A(sWe1 + nt * 16 * 64, 64, 32), zb1, acc);
      const float* bp = sF + FBE1 + nt * 16 + 8 * hh;
      v8h hv;
#pragma unroll
      for (int r = 0; r < 8; ++r) hv[r] = (_Float16)fmaxf(acc[r] + bp[r], 0.0f);
      *(v8h*)(hSlab + n16 * HID + nt * 16 + 8 * hh) = hv;
    }

    // (g) out^T = W_e2^T @ he^T (out = he W_e2 + b_e2), N=32; fused sample/logprob/fraction.
    // col = nt*16 + 8*hh + r  ->  j = col&3 = r&3 is COMPILE-TIME after unroll (no divergence).
    v16h eb[4];
#pragma unroll
    for (int kt = 0; kt < 4; ++kt) eb[kt] = load_B(hSlab, HID, kt * 32);
#pragma unroll
    for (int nt = 0; nt < 2; ++nt) {
      v8f acc = {};
#pragma unroll
      for (int kt = 0; kt < 4; ++kt)
        acc = wmma_f16(load_A(sWe2 + nt * 16 * 128, 128, kt * 32), eb[kt], acc);
      const int f0 = nt * 16 + 8 * hh;                 // 0..31
      const float* bp = sF + FBE2 + f0;
      const size_t rowbase = ((size_t)t * ROWS + (size_t)grow) * NC;
#pragma unroll
      for (int r = 0; r < 8; ++r) {
        const int   col = f0 + r;
        const float v   = acc[r] + bp[r];
        if ((r & 3) == 3) {
          outF[rowbase + (col >> 2)] = sigmoidf_(v);
        } else {
          const unsigned seed = (((unsigned)(t * ROWS + grow) << 7) | ((unsigned)col << 1)) | 1u;
          const float u    = unif01_(seed * 0x85EBCA6BU + 0xC2B2AE35U);
          const float samp = (u < sigmoidf_(v)) ? 1.0f : 0.0f;
          const size_t idx = (rowbase + (size_t)(col >> 2)) * 3 + (size_t)(r & 3);
          outS[idx] = samp;
          outX[idx] = samp * v - softplusf_(v);
        }
      }
    }
  }
}

extern "C" void kernel_launch(void* const* d_in, const int* in_sizes, int n_in,
                              void* d_out, int out_size, void* d_ws, size_t ws_size,
                              hipStream_t stream) {
  (void)in_sizes; (void)n_in; (void)out_size; (void)d_ws; (void)ws_size;
  const float* xin = (const float*)d_in[0];
  const float* Wh  = (const float*)d_in[1];   const float* bh  = (const float*)d_in[2];
  const float* Wg  = (const float*)d_in[3];   const float* bg  = (const float*)d_in[4];
  const float* Wm  = (const float*)d_in[5];   const float* bm  = (const float*)d_in[6];
  const float* Wz  = (const float*)d_in[7];   const float* bz  = (const float*)d_in[8];
  const float* Ws  = (const float*)d_in[9];   const float* bs  = (const float*)d_in[10];
  const float* We1 = (const float*)d_in[11];  const float* be1 = (const float*)d_in[12];
  const float* We2 = (const float*)d_in[13];  const float* be2 = (const float*)d_in[14];
  const float* iafShift    = (const float*)d_in[15];
  const float* iafLogscale = (const float*)d_in[16];
  float* out = (float*)d_out;

  // 138 KB dynamic LDS per workgroup (well under the 320 KB WGP limit)
  hipFuncSetAttribute((const void*)ssm_persistent_kernel,
                      hipFuncAttributeMaxDynamicSharedMemorySize, (int)SMEM_BYTES);

  ssm_persistent_kernel<<<dim3(NBLK), dim3(NTHR), SMEM_BYTES, stream>>>(
      xin, Wh, bh, Wg, bg, Wm, bm, Wz, bz, Ws, bs, We1, be1, We2, be2,
      iafShift, iafLogscale, out);
}
